// TGAT_50130858279709
// MI455X (gfx1250) — matrix operand order repaired and checked by reading, not compiled, not run on hardware
//
#include <hip/hip_runtime.h>
#include <math.h>

// ---------------- constants ----------------
static constexpr int kT  = 12;
static constexpr int kB  = 2;
static constexpr int kN  = 1024;
static constexpr int kM  = 2048;          // B*N nodes per timestep
static constexpr int kDI = 64;
static constexpr int kDH = 128;
static constexpr int kDT = 16;
static constexpr int kDO = 32;

typedef __attribute__((ext_vector_type(16))) __bf16 v16bf;
typedef __attribute__((ext_vector_type(8)))  float  v8f;

// GEMM epilogue flags (template parameter bits)
static constexpr int F_RELU     = 1;
static constexpr int F_BIAS     = 2;
static constexpr int F_ROWSCALE = 4;
static constexpr int F_PERMOUT  = 8;

// ---------------- helpers ----------------
static __device__ inline __bf16 f2bf(float f) {
  unsigned int u = __builtin_bit_cast(unsigned int, f);
  unsigned int r = u + 0x7FFFu + ((u >> 16) & 1u);   // round-to-nearest-even
  unsigned short h = (unsigned short)(r >> 16);
  return __builtin_bit_cast(__bf16, h);
}

// ---------------- small prep kernels ----------------
__global__ __launch_bounds__(256)
void mask_kernel(const unsigned char* __restrict__ ego, float* __restrict__ mf) {
  int idx = blockIdx.x * 256 + threadIdx.x;            // [T*M]
  if (idx >= kT * kM) return;
  int t = idx / kM, i = idx - t * kM;
  int b = i >> 10, n = i & 1023;
  mf[idx] = ego[(b * kT + t) * kN + n] ? 1.0f : 0.0f;  // mask.permute(1,0,2).reshape
}

__global__ __launch_bounds__(256)
void cvt_bf16_kernel(const float* __restrict__ src, __bf16* __restrict__ dst, int n) {
  int i = blockIdx.x * 256 + threadIdx.x;
  if (i < n) dst[i] = f2bf(src[i]);
}

// per-t effective Q/K/V bias: bias_eff[t] = bias + t_vec(t) @ W[128:144,:]
__global__ __launch_bounds__(128)
void biaseff_kernel(const float* __restrict__ time_w, const float* __restrict__ time_b,
                    const float* __restrict__ q_w, const float* __restrict__ q_b,
                    const float* __restrict__ k_w, const float* __restrict__ k_b,
                    const float* __restrict__ v_w, const float* __restrict__ v_b,
                    float* __restrict__ qbe, float* __restrict__ kbe, float* __restrict__ vbe) {
  int t = blockIdx.x, d = threadIdx.x;
  float aq = q_b[d], ak = k_b[d], av = v_b[d];
#pragma unroll
  for (int j = 0; j < kDT; ++j) {
    float tv = __sinf((float)t * time_w[j] + time_b[j]);
    aq += tv * q_w[(kDH + j) * kDH + d];
    ak += tv * k_w[(kDH + j) * kDH + d];
    av += tv * v_w[(kDH + j) * kDH + d];
  }
  qbe[t * kDH + d] = aq; kbe[t * kDH + d] = ak; vbe[t * kDH + d] = av;
}

// deg[i] = m[i]*(sum_j A[j,i]*m[j] + 1); nrm = deg>0 ? rsqrt(deg) : 0
__global__ __launch_bounds__(256)
void deg_kernel(const float* __restrict__ A, const float* __restrict__ mf,
                float* __restrict__ nrm) {
  __shared__ float smf[kM];
  int t = blockIdx.y;
  int i = blockIdx.x * 256 + threadIdx.x;
  for (int idx = threadIdx.x; idx < kM; idx += 256) smf[idx] = mf[(size_t)t * kM + idx];
  __syncthreads();
  const float* At = A + (size_t)t * kM * kM;
  float s = 0.f;
  for (int j = 0; j < kM; ++j) s += At[(size_t)j * kM + i] * smf[j];
  float mi = smf[i];
  float deg = mi * (s + 1.0f);
  nrm[(size_t)t * kM + i] = (deg > 0.f) ? rsqrtf(deg) : 0.f;
}

// Bmat[t][i][j] = nrm[i]*nrm[j]*A[t][j][i] + (i==j)*nrm[i]^2   (bf16)
__global__ __launch_bounds__(256)
void bmat_kernel(const float* __restrict__ A, const float* __restrict__ nrm,
                 __bf16* __restrict__ Bm) {
  __shared__ float tile[32][33];
  int t = blockIdx.z;
  int i0 = blockIdx.y * 32, j0 = blockIdx.x * 32;
  int tx = threadIdx.x, ty = threadIdx.y;              // block (32,8)
  const float* At = A + (size_t)t * kM * kM;
#pragma unroll
  for (int s = 0; s < 4; ++s) {
    int j = j0 + ty + s * 8;
    tile[ty + s * 8][tx] = At[(size_t)j * kM + i0 + tx];   // coalesced over i
  }
  __syncthreads();
  const float* nt_ = nrm + (size_t)t * kM;
  __bf16* Bt = Bm + (size_t)t * kM * kM;
#pragma unroll
  for (int s = 0; s < 4; ++s) {
    int i = i0 + ty + s * 8;
    int j = j0 + tx;
    float v = tile[tx][ty + s * 8] * nt_[i] * nt_[j];
    if (i == j) v += nt_[i] * nt_[i];
    Bt[(size_t)i * kM + j] = f2bf(v);                      // coalesced over j
  }
}

// ---------------- generic WMMA GEMM: out = f(A[t] @ B[t] + bias) ----------------
// Templated on NT (N/16) and FLAGS so accumulators stay in VGPRs (no scratch).
// A: [M x K] bf16 row-major (per-t stride aStep), B: [K x N] bf16 (per-t stride bStep)
// block = 128 threads (4 waves); wave w -> rows [bx*64 + 16w, +16), all N tiles.
template <int NT, int FLAGS>
__global__ __launch_bounds__(128)
void wmma_gemm_kernel(const __bf16* __restrict__ A, int lda, long long aStep,
                      const __bf16* __restrict__ B, int ldb, long long bStep,
                      const float* __restrict__ bias, int biasStep,
                      const float* __restrict__ rowScale,
                      void* __restrict__ outP, int ldo, long long oStep,
                      int Kdim) {
  constexpr int Ncols = NT * 16;
  __shared__ __bf16 ldsBT[Ncols * 48];                  // B tile transposed [n][k]
  const int t = blockIdx.z;
  const int tid = threadIdx.x;
  const int w = tid >> 5, L = tid & 31, L15 = L & 15, h = L >> 4;
  const int wrow0 = blockIdx.x * 64 + w * 16;
  const __bf16* At = A + (size_t)t * aStep;
  const __bf16* Bt = B + (size_t)t * bStep;

  v8f acc[NT];
#pragma unroll
  for (int i = 0; i < NT; ++i)
#pragma unroll
    for (int r = 0; r < 8; ++r) acc[i][r] = 0.f;

  const int aRow = wrow0 + L15;
  for (int kk = 0; kk < Kdim; kk += 32) {
    __syncthreads();
#pragma unroll
    for (int u = 0; u < (32 * Ncols) / 128; ++u) {
      int idx = u * 128 + tid;
      int r = idx / Ncols, c = idx - r * Ncols;
      ldsBT[c * 48 + r] = Bt[(size_t)(kk + r) * ldb + c];
    }
    __syncthreads();
    // A fragment: a[e] = A[row][kk + (e/8)*16 + h*8 + e%8] -> two 16B loads
    v16bf af;
    const __bf16* ap = At + (size_t)aRow * lda + kk + h * 8;
    *(float4*)&af        = *(const float4*)ap;
    *((float4*)&af + 1)  = *(const float4*)(ap + 16);
#pragma unroll
    for (int nt = 0; nt < NT; ++nt) {
      // B fragment: b[e] = B[kk + (e/8)*16 + h*8 + e%8][nt*16 + L15] = ldsBT[n][(e/8)*16 + h*8 + e%8]
      v16bf bf_;
      const __bf16* bp = &ldsBT[(nt * 16 + L15) * 48 + h * 8];
      *(float4*)&bf_       = *(const float4*)bp;
      *((float4*)&bf_ + 1) = *(const float4*)(bp + 16);
      acc[nt] = __builtin_amdgcn_wmma_f32_16x16x32_bf16(
          false, af, false, bf_, (short)0, acc[nt], false, false);
    }
  }

#pragma unroll
  for (int nt = 0; nt < NT; ++nt) {
    int col = nt * 16 + L15;
    float bv = (FLAGS & F_BIAS) ? bias[(size_t)t * biasStep + col] : 0.f;
#pragma unroll
    for (int r = 0; r < 8; ++r) {
      int row = wrow0 + r + 8 * h;
      float v = acc[nt][r] + bv;
      if (FLAGS & F_RELU) v = v > 0.f ? v : 0.f;
      if (FLAGS & F_ROWSCALE) v *= rowScale[(size_t)t * kM + row];
      if (FLAGS & F_PERMOUT) {           // [T,M,32] -> [B,N,T,32] fused store
        ((float*)outP)[((size_t)row * kT + t) * kDO + col] = v;
      } else {
        ((__bf16*)outP)[(size_t)t * oStep + (size_t)row * ldo + col] = f2bf(v);
      }
    }
  }
}

// ---------------- flash attention (online softmax, 32-key tiles) ----------------
__global__ __launch_bounds__(128)
void flash_attn_kernel(const __bf16* __restrict__ Qm, const __bf16* __restrict__ Kme,
                       const __bf16* __restrict__ Vm, const float* __restrict__ mf,
                       __bf16* __restrict__ AGG) {
  __shared__ __bf16 ldsK[32 * 144];        // K tile row-major [key][feat]
  __shared__ __bf16 ldsVT[128 * 48];       // V tile transposed [feat][key]
  __shared__ float  Sb[4][16][32];
  __shared__ __bf16 Pb[4][16][32];
  __shared__ float  corrS[4][16];
  __shared__ float  mfk[32];
  const int t = blockIdx.y;
  const int tid = threadIdx.x;
  const int w = tid >> 5, L = tid & 31, L15 = L & 15, h = L >> 4;
  const int wrow0 = blockIdx.x * 64 + w * 16;
  const int qrow = wrow0 + L15;

  // Q fragments resident for the whole key loop (K-dim 128 = 4 wmma steps)
  v16bf qf[4];
  const __bf16* qp = Qm + ((size_t)t * kM + qrow) * kDH;
#pragma unroll
  for (int kk = 0; kk < 4; ++kk) {
    *(float4*)&qf[kk]       = *(const float4*)(qp + kk * 32 + h * 8);
    *((float4*)&qf[kk] + 1) = *(const float4*)(qp + kk * 32 + 16 + h * 8);
  }
  v8f o[8];
#pragma unroll
  for (int i = 0; i < 8; ++i)
#pragma unroll
    for (int r = 0; r < 8; ++r) o[i][r] = 0.f;
  float mrow = -3.0e38f, lrow = 0.f;       // valid in leader lanes (L<16)

  for (int kt = 0; kt < kM / 32; ++kt) {
    const int key0 = kt * 32;
    __syncthreads();
#pragma unroll
    for (int u = 0; u < 32; ++u) {
      int idx = u * 128 + tid;
      int r = idx >> 7, c = idx & 127;
      size_t g = ((size_t)t * kM + key0 + r) * kDH + c;
      ldsK[r * 144 + c]  = Kme[g];
      ldsVT[c * 48 + r]  = Vm[g];
    }
    if (tid < 32) mfk[tid] = mf[(size_t)t * kM + key0 + tid];
    __syncthreads();

    // S = Q @ K^T : b[e] = K[key][kk*32 + (e/8)*16 + h*8 + e%8]
    v8f s0, s1;
#pragma unroll
    for (int r = 0; r < 8; ++r) { s0[r] = 0.f; s1[r] = 0.f; }
#pragma unroll
    for (int kk = 0; kk < 4; ++kk) {
      v16bf b0, b1;
      const __bf16* p0 = &ldsK[L15 * 144 + kk * 32 + h * 8];
      const __bf16* p1 = &ldsK[(16 + L15) * 144 + kk * 32 + h * 8];
      *(float4*)&b0 = *(const float4*)p0; *((float4*)&b0 + 1) = *(const float4*)(p0 + 16);
      *(float4*)&b1 = *(const float4*)p1; *((float4*)&b1 + 1) = *(const float4*)(p1 + 16);
      s0 = __builtin_amdgcn_wmma_f32_16x16x32_bf16(false, qf[kk], false, b0, (short)0, s0, false, false);
      s1 = __builtin_amdgcn_wmma_f32_16x16x32_bf16(false, qf[kk], false, b1, (short)0, s1, false, false);
    }
    const float scale = 0.088388347648318447f;   // 1/sqrt(128)
#pragma unroll
    for (int r = 0; r < 8; ++r) {
      Sb[w][r + 8 * h][L15]      = s0[r] * scale;
      Sb[w][r + 8 * h][16 + L15] = s1[r] * scale;
    }
    asm volatile("s_wait_dscnt 0" ::: "memory");   // wave-lockstep cross-lane LDS

    if (L < 16) {                                  // leader lane L owns row L
      float tmax = -3.0e38f, sv[32];
#pragma unroll
      for (int j = 0; j < 32; ++j) {
        float xv = Sb[w][L][j];
        xv = (mfk[j] != 0.f) ? xv : -1.0e9f;       // key mask
        sv[j] = xv; tmax = fmaxf(tmax, xv);
      }
      float mnew = fmaxf(mrow, tmax);
      float corrv = __expf(mrow - mnew);
      float ssum = 0.f;
#pragma unroll
      for (int j = 0; j < 32; ++j) {
        float p = __expf(sv[j] - mnew);
        ssum += p;
        Pb[w][L][j] = f2bf(p);
      }
      lrow = lrow * corrv + ssum;
      mrow = mnew;
      corrS[w][L] = corrv;
    }
    asm volatile("s_wait_dscnt 0" ::: "memory");

    float cf[8];
#pragma unroll
    for (int r = 0; r < 8; ++r) cf[r] = corrS[w][r + 8 * h];
#pragma unroll
    for (int nt = 0; nt < 8; ++nt)
#pragma unroll
      for (int r = 0; r < 8; ++r) o[nt][r] *= cf[r];

    // P fragment (16x32 A operand) from LDS
    v16bf pa;
    *(float4*)&pa       = *(const float4*)&Pb[w][L15][h * 8];
    *((float4*)&pa + 1) = *(const float4*)&Pb[w][L15][16 + h * 8];
#pragma unroll
    for (int nt = 0; nt < 8; ++nt) {
      v16bf bv;
      const __bf16* vp = &ldsVT[(nt * 16 + L15) * 48 + h * 8];
      *(float4*)&bv       = *(const float4*)vp;
      *((float4*)&bv + 1) = *(const float4*)(vp + 16);
      o[nt] = __builtin_amdgcn_wmma_f32_16x16x32_bf16(false, pa, false, bv, (short)0, o[nt], false, false);
    }
  }

  if (L < 16) corrS[w][L] = 1.0f / lrow;
  asm volatile("s_wait_dscnt 0" ::: "memory");
  float inv[8];
#pragma unroll
  for (int r = 0; r < 8; ++r) inv[r] = corrS[w][r + 8 * h];
#pragma unroll
  for (int nt = 0; nt < 8; ++nt)
#pragma unroll
    for (int r = 0; r < 8; ++r) {
      int row = wrow0 + r + 8 * h;
      AGG[((size_t)t * kM + row) * kDH + nt * 16 + L15] = f2bf(o[nt][r] * inv[r]);
    }
}

// ---------------- host orchestration ----------------
extern "C" void kernel_launch(void* const* d_in, const int* in_sizes, int n_in,
                              void* d_out, int out_size, void* d_ws, size_t ws_size,
                              hipStream_t stream) {
  const float* x       = (const float*)d_in[0];
  const float* A       = (const float*)d_in[1];
  const unsigned char* ego = (const unsigned char*)d_in[2];
  const float* gcn1_w  = (const float*)d_in[3];
  const float* gcn1_b  = (const float*)d_in[4];
  const float* gcn2_w  = (const float*)d_in[5];
  const float* gcn2_b  = (const float*)d_in[6];
  const float* time_w  = (const float*)d_in[7];
  const float* time_b  = (const float*)d_in[8];
  const float* q_w     = (const float*)d_in[9];
  const float* q_b     = (const float*)d_in[10];
  const float* k_w     = (const float*)d_in[11];
  const float* k_b     = (const float*)d_in[12];
  const float* v_w     = (const float*)d_in[13];
  const float* v_b     = (const float*)d_in[14];
  const float* o_w     = (const float*)d_in[15];
  const float* o_b     = (const float*)d_in[16];
  const float* fc_w    = (const float*)d_in[17];
  const float* fc_b    = (const float*)d_in[18];
  float* out           = (float*)d_out;

  char* ws = (char*)d_ws;
  size_t off = 0;
  auto alloc = [&](size_t bytes) -> void* {
    void* p = ws + off;
    off = (off + bytes + 255) & ~(size_t)255;
    return p;
  };
  float*  mf   = (float*) alloc((size_t)kT * kM * 4);
  float*  nrm  = (float*) alloc((size_t)kT * kM * 4);
  __bf16* xbf  = (__bf16*)alloc((size_t)kT * kM * kDI * 2);
  __bf16* w1b  = (__bf16*)alloc((size_t)kDI * kDH * 2);
  __bf16* w2b  = (__bf16*)alloc((size_t)kDH * kDH * 2);
  __bf16* qwb  = (__bf16*)alloc((size_t)(kDH + kDT) * kDH * 2);
  __bf16* kwb  = (__bf16*)alloc((size_t)(kDH + kDT) * kDH * 2);
  __bf16* vwb  = (__bf16*)alloc((size_t)(kDH + kDT) * kDH * 2);
  __bf16* owb  = (__bf16*)alloc((size_t)kDH * kDH * 2);
  __bf16* fcwb = (__bf16*)alloc((size_t)kDH * kDO * 2);
  float*  qbe  = (float*) alloc((size_t)kT * kDH * 4);
  float*  kbe  = (float*) alloc((size_t)kT * kDH * 4);
  float*  vbe  = (float*) alloc((size_t)kT * kDH * 4);
  __bf16* Bmat = (__bf16*)alloc((size_t)kT * kM * kM * 2);
  __bf16* PA   = (__bf16*)alloc((size_t)kT * kM * kDH * 2);
  __bf16* PB   = (__bf16*)alloc((size_t)kT * kM * kDH * 2);
  __bf16* Qm   = (__bf16*)alloc((size_t)kT * kM * kDH * 2);
  __bf16* Km   = (__bf16*)alloc((size_t)kT * kM * kDH * 2);
  __bf16* Vm   = (__bf16*)alloc((size_t)kT * kM * kDH * 2);

  const long long sMH = (long long)kM * kDH;   // per-t stride of [M,128] bf16
  const long long sMM = (long long)kM * kM;

  // prep
  mask_kernel<<<(kT * kM + 255) / 256, 256, 0, stream>>>(ego, mf);
  auto cvt = [&](const float* s, __bf16* d, int n) {
    cvt_bf16_kernel<<<(n + 255) / 256, 256, 0, stream>>>(s, d, n);
  };
  cvt(x, xbf, kT * kM * kDI);
  cvt(gcn1_w, w1b, kDI * kDH);
  cvt(gcn2_w, w2b, kDH * kDH);
  cvt(q_w, qwb, (kDH + kDT) * kDH);
  cvt(k_w, kwb, (kDH + kDT) * kDH);
  cvt(v_w, vwb, (kDH + kDT) * kDH);
  cvt(o_w, owb, kDH * kDH);
  cvt(fc_w, fcwb, kDH * kDO);
  biaseff_kernel<<<kT, kDH, 0, stream>>>(time_w, time_b, q_w, q_b, k_w, k_b,
                                         v_w, v_b, qbe, kbe, vbe);
  deg_kernel<<<dim3(kM / 256, kT), 256, 0, stream>>>(A, mf, nrm);
  bmat_kernel<<<dim3(kM / 32, kM / 32, kT), dim3(32, 8), 0, stream>>>(A, nrm, Bmat);

  dim3 gg(kM / 64, 1, kT);
  // P = x @ W1                         [2048x64]@[64x128]
  wmma_gemm_kernel<8, 0><<<gg, 128, 0, stream>>>(xbf, kDI, (long long)kM * kDI,
      w1b, kDH, 0, nullptr, 0, nullptr, PA, kDH, sMH, kDI);
  // H1 = relu(Bmat @ P + b1)           [2048x2048]@[2048x128]
  wmma_gemm_kernel<8, F_RELU | F_BIAS><<<gg, 128, 0, stream>>>(Bmat, kM, sMM,
      PA, kDH, sMH, gcn1_b, 0, nullptr, PB, kDH, sMH, kM);
  // P = H1 @ W2
  wmma_gemm_kernel<8, 0><<<gg, 128, 0, stream>>>(PB, kDH, sMH,
      w2b, kDH, 0, nullptr, 0, nullptr, PA, kDH, sMH, kDH);
  // H2 = relu(Bmat @ P + b2)
  wmma_gemm_kernel<8, F_RELU | F_BIAS><<<gg, 128, 0, stream>>>(Bmat, kM, sMM,
      PA, kDH, sMH, gcn2_b, 0, nullptr, PB, kDH, sMH, kM);
  // Q/K/V = H2 @ W[0:128] + bias_eff[t]  (time-embed folded into bias)
  wmma_gemm_kernel<8, F_BIAS><<<gg, 128, 0, stream>>>(PB, kDH, sMH,
      qwb, kDH, 0, qbe, kDH, nullptr, Qm, kDH, sMH, kDH);
  wmma_gemm_kernel<8, F_BIAS><<<gg, 128, 0, stream>>>(PB, kDH, sMH,
      kwb, kDH, 0, kbe, kDH, nullptr, Km, kDH, sMH, kDH);
  wmma_gemm_kernel<8, F_BIAS><<<gg, 128, 0, stream>>>(PB, kDH, sMH,
      vwb, kDH, 0, vbe, kDH, nullptr, Vm, kDH, sMH, kDH);
  // AGG = softmax(QK^T/sqrt(128), key-masked) @ V   -> PA
  flash_attn_kernel<<<dim3(kM / 64, kT), 128, 0, stream>>>(Qm, Km, Vm, mf, PA);
  // AG2 = AGG @ o_w + o_b  -> Km (reuse)
  wmma_gemm_kernel<8, F_BIAS><<<gg, 128, 0, stream>>>(PA, kDH, sMH,
      owb, kDH, 0, o_b, 0, nullptr, Km, kDH, sMH, kDH);
  // out = perm((AG2 @ fc_w + fc_b) * m)  fused [B,N,T,32] store
  wmma_gemm_kernel<2, F_BIAS | F_ROWSCALE | F_PERMOUT><<<gg, 128, 0, stream>>>(
      Km, kDH, sMH, fcwb, kDO, 0, fc_b, 0, mf, out, kDO, 0, kDH);

  (void)in_sizes; (void)n_in; (void)out_size; (void)ws_size;
}
